// SelfAttention_3994319585491
// MI455X (gfx1250) — matrix operand order repaired and checked by reading, not compiled, not run on hardware
//
#include <hip/hip_runtime.h>
#include <hip/hip_bf16.h>
#include <stdint.h>

// ---------------------------------------------------------------------------
// Self-attention forward on MI455X (gfx1250).
//  - All matmuls: v_wmma_f32_16x16x32_bf16 (wave32 WMMA, fp32 accumulate)
//  - GEMM operand staging: Tensor Data Mover (tensor_load_to_lds, 6-arg
//    builtin on this toolchain) into double-buffered LDS, synchronized with
//    s_wait_tensorcnt; cooperative-load fallback if the builtin is absent.
//  - Attention: flash-style S^T = K·Q^T formulation, online softmax with
//    lane-local reductions, P repacked to the WMMA A-fragment in registers.
// ---------------------------------------------------------------------------

#define NUM_HEADS 16
#define HEAD_DIM  64
#define EMB       1024
#define SEQ       2048
#define BATCH     4
#define MROWS     (BATCH * SEQ)   // 8192
#define KSTAGE    64              // K-extent staged in LDS per buffer

typedef unsigned short bf16_t;
typedef __attribute__((ext_vector_type(16))) __bf16 v16bf;
typedef __attribute__((ext_vector_type(8)))  float  v8f;
typedef __attribute__((ext_vector_type(4)))  unsigned int u32x4;
typedef __attribute__((ext_vector_type(8)))  int  i32x8;
typedef __attribute__((ext_vector_type(4)))  int  i32x4;

struct Frag2 { uint4 lo, hi; };            // 32 bytes == one v16bf
struct H16   { unsigned short h[16]; };    // 32 bytes == one v16bf
struct H8    { unsigned short h[8]; };     // 16 bytes == one uint4

#if defined(__gfx1250__) && __has_builtin(__builtin_amdgcn_tensor_load_to_lds) && \
    __has_builtin(__builtin_amdgcn_s_wait_tensorcnt)
#define USE_TDM 1
#endif

__device__ __forceinline__ unsigned short f32_bf16(float f) {
    unsigned int u = __builtin_bit_cast(unsigned int, f);
    u += 0x7FFFu + ((u >> 16) & 1u);       // round-to-nearest-even
    return (unsigned short)(u >> 16);
}

// A-operand fragment (16x32, M x K). 16-bit A layout:
//   lane l<16 : row l, K = {k0..k0+7, k0+16..k0+23}
//   lane l+16 : row l, K = {k0+8..k0+15, k0+24..k0+31}
__device__ __forceinline__ v16bf load_a_frag(const bf16_t* __restrict__ base,
                                             int ld, int row0, int k0) {
    int lane = threadIdx.x & 31;
    const bf16_t* p = base + (size_t)(row0 + (lane & 15)) * ld
                           + k0 + ((lane >> 4) << 3);
    Frag2 f;
    f.lo = *(const uint4*)p;          // 8 bf16
    f.hi = *(const uint4*)(p + 16);   // 8 bf16, +32B
    return __builtin_bit_cast(v16bf, f);
}

// B-operand fragment (32x16, K x N), column n contiguous in memory (ld elems):
//   lane n<16 : col n, K = {k0..k0+7, k0+16..k0+23} ; lane n+16 : col n, K = {k0+8..k0+15, k0+24..k0+31}
__device__ __forceinline__ v16bf load_b_frag(const bf16_t* __restrict__ base,
                                             int ld, int col0, int k0) {
    int lane = threadIdx.x & 31;
    const bf16_t* p = base + (size_t)(col0 + (lane & 15)) * ld
                           + k0 + ((lane >> 4) << 3);
    Frag2 f;
    f.lo = *(const uint4*)p;
    f.hi = *(const uint4*)(p + 16);
    return __builtin_bit_cast(v16bf, f);
}

__device__ __forceinline__ v8f wmma_bf16(v16bf a, v16bf b, v8f c) {
    return __builtin_amdgcn_wmma_f32_16x16x32_bf16(
        false, a, false, b, (short)0, c, false, false);
}

#ifdef USE_TDM
// Issue a TDM 2-D tile load: tile_d1 rows x tile_d0 elements (bf16), row
// stride `stride0` elements, packed row-major into LDS at byte offset
// lds_addr. D# fields per CDNA5 ISA (groups 0..3); this
// toolchain's builtin takes 6 args: (g0, g1, g2, g3, g4, cpol).
__device__ __forceinline__ void tdm_load_2d(unsigned lds_addr, const void* gptr,
                                            unsigned tensor_d0, unsigned tensor_d1,
                                            unsigned tile_d0, unsigned tile_d1,
                                            unsigned stride0) {
    unsigned long long ga = (unsigned long long)(uintptr_t)gptr;
    u32x4 g0;
    g0[0] = 1u;                                   // count=1, user mode
    g0[1] = lds_addr;                             // lds_addr [63:32]
    g0[2] = (unsigned)ga;                         // global_addr [95:64]
    g0[3] = (unsigned)((ga >> 32) & 0x01FFFFFFu)  // global_addr [120:96]
          | (2u << 30);                           // type=2 ("image")
    i32x8 g1;
    g1[0] = (int)(1u << 16);                      // data_size=1 (2 bytes)
    g1[1] = (int)((tensor_d0 & 0xFFFFu) << 16);   // tensor_dim0[15:0] @ bit48
    g1[2] = (int)(((tensor_d0 >> 16) & 0xFFFFu)   // tensor_dim0[31:16]
          | ((tensor_d1 & 0xFFFFu) << 16));       // tensor_dim1[15:0]
    g1[3] = (int)(((tensor_d1 >> 16) & 0xFFFFu)   // tensor_dim1[31:16]
          | (tile_d0 << 16));                     // tile_dim0 @ bit112
    g1[4] = (int)(tile_d1 & 0xFFFFu);             // tile_dim1; tile_dim2=0
    g1[5] = (int)stride0;                         // tensor_dim0_stride[31:0]
    g1[6] = 0;                                    // stride hi / dim1_stride
    g1[7] = 0;
    i32x4 gz;
    gz[0] = 0; gz[1] = 0; gz[2] = 0; gz[3] = 0;   // 2-D: groups 2/3 unused
    i32x8 gz8;
    gz8[0] = 0; gz8[1] = 0; gz8[2] = 0; gz8[3] = 0;
    gz8[4] = 0; gz8[5] = 0; gz8[6] = 0; gz8[7] = 0;
    __builtin_amdgcn_tensor_load_to_lds(g0, g1, gz, gz, gz8, 0);
}
#endif

// ---------------------------------------------------------------------------
__global__ void cvt_f32_bf16(const float* __restrict__ in,
                             bf16_t* __restrict__ out, int n) {
    for (int i = blockIdx.x * blockDim.x + threadIdx.x; i < n;
         i += gridDim.x * blockDim.x)
        out[i] = f32_bf16(in[i]);
}

#ifndef USE_TDM
// Fallback: cooperative copy of rows x KSTAGE bf16 (row stride ld) into LDS.
__device__ __forceinline__ void coop_copy(bf16_t* __restrict__ dst,
                                          const bf16_t* __restrict__ src,
                                          int ld, int rows) {
    const int chunks = rows * (KSTAGE / 8);       // 16B chunks
    for (int c = threadIdx.x; c < chunks; c += blockDim.x) {
        int r = c >> 3, kc = (c & 7) << 3;
        *(uint4*)(dst + r * KSTAGE + kc) = *(const uint4*)(src + (size_t)r * ld + kc);
    }
}
#endif

// ---------------------------------------------------------------------------
// C[M,N] = A[M,K] * W[N,K]^T + bias, bf16 in, fp32 accumulate.
// Block: 128 threads (4 waves), tile 128(M) x 64(N); K staged 64-wide,
// double-buffered in LDS via TDM. Wave tile 32x64 -> 8 accumulators.
// mode 0: fp32 out[M,N]
// mode 1: bf16 [B,H,S,64]  (value * scale)   -- Q / K layout
// mode 2: bf16 [B,H,64,S]  (transposed)      -- V^T layout, packed stores
__global__ void gemm_xwt_wmma(const bf16_t* __restrict__ A,
                              const bf16_t* __restrict__ W,
                              const float* __restrict__ bias,
                              void* __restrict__ out,
                              int K, int N, int mode, float scale) {
    __shared__ alignas(16) bf16_t As[2][128 * KSTAGE];   // 2 x 16 KB
    __shared__ alignas(16) bf16_t Bs[2][64 * KSTAGE];    // 2 x  8 KB

    const int lane  = threadIdx.x & 31;
    const int wave  = threadIdx.x >> 5;
    const int n0    = blockIdx.x * 64;
    const int mblk  = blockIdx.y * 128;
    const int m0    = mblk + wave * 32;
    const int half8 = (lane >> 4) << 3;
    const int colL  = lane & 15;
    const int nstage = K / KSTAGE;

    const bf16_t* Ab = A + (size_t)mblk * K;
    const bf16_t* Wb = W + (size_t)n0 * K;

    v8f acc[2][4];
#pragma unroll
    for (int i = 0; i < 2; ++i)
#pragma unroll
        for (int j = 0; j < 4; ++j)
#pragma unroll
            for (int e = 0; e < 8; ++e) acc[i][j][e] = 0.f;

#ifdef USE_TDM
    if (wave == 0) {   // prologue: DMA stage 0 into buffer 0
        tdm_load_2d((unsigned)(uintptr_t)&As[0][0], Ab, K, MROWS, KSTAGE, 128, K);
        tdm_load_2d((unsigned)(uintptr_t)&Bs[0][0], Wb, K, EMB,   KSTAGE, 64,  K);
    }
#else
    coop_copy(&As[0][0], Ab, K, 128);
    coop_copy(&Bs[0][0], Wb, K, 64);
#endif

    for (int st = 0; st < nstage; ++st) {
        const int cur = st & 1;
#ifdef USE_TDM
        if (wave == 0) {
            if (st + 1 < nstage) {   // DMA next stage into other buffer
                const bf16_t* An = Ab + (st + 1) * KSTAGE;
                const bf16_t* Wn = Wb + (st + 1) * KSTAGE;
                tdm_load_2d((unsigned)(uintptr_t)&As[cur ^ 1][0], An, K, MROWS,
                            KSTAGE, 128, K);
                tdm_load_2d((unsigned)(uintptr_t)&Bs[cur ^ 1][0], Wn, K, EMB,
                            KSTAGE, 64, K);
                // in-order per-wave completion: <=2 outstanding => cur done
                __builtin_amdgcn_s_wait_tensorcnt(2);
            } else {
                __builtin_amdgcn_s_wait_tensorcnt(0);
            }
        }
#endif
        __syncthreads();   // staged data visible to all waves

        const bf16_t* Ac = &As[cur][0];
        const bf16_t* Bc = &Bs[cur][0];
#pragma unroll
        for (int kk = 0; kk < KSTAGE; kk += 32) {
            v16bf a0 = load_a_frag(Ac, KSTAGE, wave * 32,      kk);
            v16bf a1 = load_a_frag(Ac, KSTAGE, wave * 32 + 16, kk);
            v16bf b0 = load_b_frag(Bc, KSTAGE, 0,  kk);
            v16bf b1 = load_b_frag(Bc, KSTAGE, 16, kk);
            v16bf b2 = load_b_frag(Bc, KSTAGE, 32, kk);
            v16bf b3 = load_b_frag(Bc, KSTAGE, 48, kk);
            acc[0][0] = wmma_bf16(a0, b0, acc[0][0]);
            acc[0][1] = wmma_bf16(a0, b1, acc[0][1]);
            acc[0][2] = wmma_bf16(a0, b2, acc[0][2]);
            acc[0][3] = wmma_bf16(a0, b3, acc[0][3]);
            acc[1][0] = wmma_bf16(a1, b0, acc[1][0]);
            acc[1][1] = wmma_bf16(a1, b1, acc[1][1]);
            acc[1][2] = wmma_bf16(a1, b2, acc[1][2]);
            acc[1][3] = wmma_bf16(a1, b3, acc[1][3]);
        }
        __syncthreads();   // all reads of `cur` done before it is overwritten
#ifndef USE_TDM
        if (st + 1 < nstage) {
            coop_copy(&As[cur ^ 1][0], Ab + (st + 1) * KSTAGE, K, 128);
            coop_copy(&Bs[cur ^ 1][0], Wb + (st + 1) * KSTAGE, K, 64);
        }
#endif
    }

    // Epilogue. C layout: VGPR e <-> local row e + half8, lane&15 <-> local col.
#pragma unroll
    for (int i = 0; i < 2; ++i) {
#pragma unroll
        for (int j = 0; j < 4; ++j) {
            const int n  = n0 + 16 * j + colL;
            const float bv = bias[n];
            if (mode == 0) {
                float* o = (float*)out;
#pragma unroll
                for (int e = 0; e < 8; ++e) {
                    int m = m0 + 16 * i + half8 + e;
                    o[(size_t)m * N + n] = acc[i][j][e] + bv;
                }
            } else if (mode == 1) {
                bf16_t* o = (bf16_t*)out;
                const int h = n >> 6, d = n & 63;
#pragma unroll
                for (int e = 0; e < 8; ++e) {
                    int m = m0 + 16 * i + half8 + e;
                    int b = m >> 11, s = m & (SEQ - 1);
                    size_t idx = ((((size_t)b * NUM_HEADS + h) * SEQ + s) << 6) + d;
                    o[idx] = f32_bf16((acc[i][j][e] + bv) * scale);
                }
            } else {
                // V^T [B,H,64,S]: 8 row-elements are consecutive s -> pack 16B
                bf16_t* o = (bf16_t*)out;
                const int h = n >> 6, d = n & 63;
                int ms = m0 + 16 * i + half8;
                int b = ms >> 11, s = ms & (SEQ - 1);
                H8 pk;
#pragma unroll
                for (int e = 0; e < 8; ++e)
                    pk.h[e] = f32_bf16(acc[i][j][e] + bv);
                size_t idx = (((size_t)b * NUM_HEADS + h) * HEAD_DIM + d) * SEQ + s;
                *(uint4*)(o + idx) = __builtin_bit_cast(uint4, pk);
            }
        }
    }
}

// ---------------------------------------------------------------------------
// Flash-style attention, one wave per 16-query tile, no LDS, no transposes.
__global__ void attn_wmma(const bf16_t* __restrict__ Qb,
                          const bf16_t* __restrict__ Kb,
                          const bf16_t* __restrict__ VTb,
                          bf16_t* __restrict__ ctx) {
    const int lane = threadIdx.x & 31;
    const int wave = threadIdx.x >> 5;
    const int q0   = blockIdx.x * 64 + wave * 16;
    const int bh   = blockIdx.y;

    const bf16_t* Qp = Qb  + (size_t)bh * SEQ * HEAD_DIM;
    const bf16_t* Kp = Kb  + (size_t)bh * SEQ * HEAD_DIM;
    const bf16_t* Vp = VTb + (size_t)bh * HEAD_DIM * SEQ;

    // Q as B-operand (contraction d=64 -> two frags), held for the whole loop.
    v16bf qf0 = load_b_frag(Qp, HEAD_DIM, q0, 0);
    v16bf qf1 = load_b_frag(Qp, HEAD_DIM, q0, 32);

    v8f o[4];
#pragma unroll
    for (int j = 0; j < 4; ++j)
#pragma unroll
        for (int e = 0; e < 8; ++e) o[j][e] = 0.f;

    float m_run = -1e30f, l_run = 0.f;
    const int scSel = (lane & 16) >> 1;   // +8 for high-half lanes

    for (int kv = 0; kv < SEQ; kv += 32) {
        v8f s0, s1;
#pragma unroll
        for (int e = 0; e < 8; ++e) { s0[e] = 0.f; s1[e] = 0.f; }

        v16bf ka0 = load_a_frag(Kp, HEAD_DIM, kv,      0);
        v16bf ka1 = load_a_frag(Kp, HEAD_DIM, kv,      32);
        s0 = wmma_bf16(ka0, qf0, s0);
        s0 = wmma_bf16(ka1, qf1, s0);
        v16bf kb0 = load_a_frag(Kp, HEAD_DIM, kv + 16, 0);
        v16bf kb1 = load_a_frag(Kp, HEAD_DIM, kv + 16, 32);
        s1 = wmma_bf16(kb0, qf0, s1);
        s1 = wmma_bf16(kb1, qf1, s1);

        // Online softmax along kv (rows of S^T): lane-local + one xor-16.
        float mx = -1e30f;
#pragma unroll
        for (int e = 0; e < 8; ++e) mx = fmaxf(mx, fmaxf(s0[e], s1[e]));
        mx = fmaxf(mx, __shfl_xor(mx, 16, 32));
        const float m_new = fmaxf(m_run, mx);
        const float corr  = __expf(m_run - m_new);

        float rs = 0.f;
        H16 ph;
#pragma unroll
        for (int e = 0; e < 8; ++e) {
            float t = __expf(s0[e] - m_new);
            rs += t; ph.h[e] = f32_bf16(t);
        }
#pragma unroll
        for (int e = 0; e < 8; ++e) {
            float t = __expf(s1[e] - m_new);
            rs += t; ph.h[8 + e] = f32_bf16(t);
        }
        rs += __shfl_xor(rs, 16, 32);
        l_run = l_run * corr + rs;
        m_run = m_new;

        // P as A-operand: exactly the two stacked C tiles -> pure bit_cast.
        v16bf pa = __builtin_bit_cast(v16bf, ph);

        // Rescale O (q lives in VGPR index here; fetch corr for that q).
#pragma unroll
        for (int e = 0; e < 8; ++e) {
            float c = __shfl(corr, e + scSel, 32);
#pragma unroll
            for (int j = 0; j < 4; ++j) o[j][e] *= c;
        }

        v16bf v0 = load_b_frag(Vp, SEQ, 0,  kv);
        v16bf v1 = load_b_frag(Vp, SEQ, 16, kv);
        v16bf v2 = load_b_frag(Vp, SEQ, 32, kv);
        v16bf v3 = load_b_frag(Vp, SEQ, 48, kv);
        o[0] = wmma_bf16(pa, v0, o[0]);
        o[1] = wmma_bf16(pa, v1, o[1]);
        o[2] = wmma_bf16(pa, v2, o[2]);
        o[3] = wmma_bf16(pa, v3, o[3]);
    }

    // Finalize: divide by row sum, store bf16 context [B,S,D].
    const int h = bh & (NUM_HEADS - 1);
    const int b = bh >> 4;
#pragma unroll
    for (int e = 0; e < 8; ++e) {
        float li  = __shfl(l_run, e + scSel, 32);
        float inv = 1.0f / li;
        int s = q0 + ((lane >> 4) << 3) + e;
        size_t row = ((size_t)b * SEQ + s) * EMB + h * HEAD_DIM;
#pragma unroll
        for (int j = 0; j < 4; ++j)
            ctx[row + 16 * j + (lane & 15)] = f32_bf16(o[j][e] * inv);
    }
}

// ---------------------------------------------------------------------------
extern "C" void kernel_launch(void* const* d_in, const int* in_sizes, int n_in,
                              void* d_out, int out_size, void* d_ws, size_t ws_size,
                              hipStream_t stream) {
    const float* x  = (const float*)d_in[0];
    const float* Wq = (const float*)d_in[1];
    const float* bq = (const float*)d_in[2];
    const float* Wk = (const float*)d_in[3];
    const float* bk = (const float*)d_in[4];
    const float* Wv = (const float*)d_in[5];
    const float* bv = (const float*)d_in[6];
    const float* Wo = (const float*)d_in[7];
    const float* bo = (const float*)d_in[8];
    float* out = (float*)d_out;

    // Workspace carve-up (bf16 elements); total ~88 MB.
    bf16_t* p    = (bf16_t*)d_ws;
    bf16_t* xb   = p;                 p += (size_t)MROWS * EMB;   // 16 MB
    bf16_t* wqb  = p;                 p += (size_t)EMB * EMB;     //  2 MB
    bf16_t* wkb  = p;                 p += (size_t)EMB * EMB;
    bf16_t* wvb  = p;                 p += (size_t)EMB * EMB;
    bf16_t* wob  = p;                 p += (size_t)EMB * EMB;
    bf16_t* Qbuf = p;                 p += (size_t)MROWS * EMB;   // [B,H,S,64]
    bf16_t* Kbuf = p;                 p += (size_t)MROWS * EMB;   // [B,H,S,64]
    bf16_t* VTb  = p;                 p += (size_t)MROWS * EMB;   // [B,H,64,S]
    bf16_t* ctxb = p;                 p += (size_t)MROWS * EMB;   // [B,S,D]

    // 1) bf16 conversion
    cvt_f32_bf16<<<2048, 256, 0, stream>>>(x,  xb,  MROWS * EMB);
    cvt_f32_bf16<<<1024, 256, 0, stream>>>(Wq, wqb, EMB * EMB);
    cvt_f32_bf16<<<1024, 256, 0, stream>>>(Wk, wkb, EMB * EMB);
    cvt_f32_bf16<<<1024, 256, 0, stream>>>(Wv, wvb, EMB * EMB);
    cvt_f32_bf16<<<1024, 256, 0, stream>>>(Wo, wob, EMB * EMB);

    // 2) QKV projections (grid: N/64 x M/128, 4 waves/block)
    dim3 ggrid(EMB / 64, MROWS / 128);
    const float qscale = 0.125f;  // 1/sqrt(HEAD_DIM)
    gemm_xwt_wmma<<<ggrid, 128, 0, stream>>>(xb, wqb, bq, Qbuf, EMB, EMB, 1, qscale);
    gemm_xwt_wmma<<<ggrid, 128, 0, stream>>>(xb, wkb, bk, Kbuf, EMB, EMB, 1, 1.0f);
    gemm_xwt_wmma<<<ggrid, 128, 0, stream>>>(xb, wvb, bv, VTb,  EMB, EMB, 2, 1.0f);

    // 3) attention: grid (S/64, B*H), 4 waves/block, 16 queries per wave
    attn_wmma<<<dim3(SEQ / 64, BATCH * NUM_HEADS), 128, 0, stream>>>(
        Qbuf, Kbuf, VTb, ctxb);

    // 4) output projection, fp32 result
    gemm_xwt_wmma<<<ggrid, 128, 0, stream>>>(ctxb, wob, bo, out, EMB, EMB, 0, 1.0f);
}
